// TranEmbeder_31129922961867
// MI455X (gfx1250) — matrix operand order repaired and checked by reading, not compiled, not run on hardware
//
#include <hip/hip_runtime.h>
#include <hip/hip_bf16.h>
#include <cstdint>
#include <cstddef>

// ---------------------------------------------------------------------------
// Model dims (fixed by the reference)
// ---------------------------------------------------------------------------
constexpr int kH  = 512;    // hidden
constexpr int kE  = 512;    // embedding
constexpr int kG4 = 2048;   // 4*H gate width

typedef __attribute__((ext_vector_type(16))) __bf16 v16bf;
typedef __attribute__((ext_vector_type(8)))  __bf16 v8bf;
typedef __attribute__((ext_vector_type(8)))  float  v8f;

__device__ __forceinline__ unsigned short f2bf(float x) {
  unsigned int u = __float_as_uint(x);
  return (unsigned short)((u + 0x7FFFu + ((u >> 16) & 1u)) >> 16);  // RNE
}
__device__ __forceinline__ float bf2f(unsigned short x) {
  return __uint_as_float(((unsigned int)x) << 16);
}
__device__ __forceinline__ float sigmoidf_(float x) {
  return 1.0f / (1.0f + __expf(-x));
}

// ---------------------------------------------------------------------------
// f32 -> bf16 weight conversion (vectorized float4 -> ushort4)
// ---------------------------------------------------------------------------
__global__ void cvt_bf16_kernel(const float* __restrict__ src,
                                unsigned short* __restrict__ dst, int n4) {
  int i = blockIdx.x * blockDim.x + threadIdx.x;
  if (i >= n4) return;
  float4 v = reinterpret_cast<const float4*>(src)[i];
  ushort4 o;
  o.x = f2bf(v.x); o.y = f2bf(v.y); o.z = f2bf(v.z); o.w = f2bf(v.w);
  reinterpret_cast<ushort4*>(dst)[i] = o;
}

// ---------------------------------------------------------------------------
// Embedding gather -> time-major bf16 activations  X[t*NS + n, :] = emb[tok]
// One block per row, 128 threads x 4 elems (E = 512).
// Sequences 0..NS1-1 come from tokA, NS1..NS-1 from tokB.
// ---------------------------------------------------------------------------
__global__ void gather_embed_kernel(const int* __restrict__ tokA,
                                    const int* __restrict__ tokB,
                                    int NS1, int NS, int T,
                                    const float* __restrict__ emb,
                                    unsigned short* __restrict__ X) {
  int row = blockIdx.x;            // row = t*NS + n
  int t = row / NS;
  int n = row - t * NS;
  int tok = (n < NS1) ? tokA[n * T + t] : tokB[(n - NS1) * T + t];
  const float4* src = reinterpret_cast<const float4*>(emb + (size_t)tok * kE);
  ushort4* dst = reinterpret_cast<ushort4*>(X + (size_t)row * kE);
  int e = threadIdx.x;             // 0..127
  float4 v = src[e];
  ushort4 o;
  o.x = f2bf(v.x); o.y = f2bf(v.y); o.z = f2bf(v.z); o.w = f2bf(v.w);
  dst[e] = o;
}

// ---------------------------------------------------------------------------
// WMMA bf16 GEMM:  C[M,Nc] = A[M,K] * Bt[Nc,K]^T  (+ Cin_bf16) (+ bias) (tanh)
// A, Bt row-major bf16 with contiguous K.
// Register-blocked: each wave32 owns a 32x64 macro-tile = 2(M) x 4(N) WMMA
// tiles -> 8 x v_wmma_f32_16x16x32_bf16 per K-step, A frags reused 4x,
// B frags reused 2x (1.5 b128 loads per WMMA instead of 4).
// 8 waves / 256-thread block; consecutive waves take consecutive N tiles
// for L0/L2 locality on the (L2-resident) weight matrix.
// ---------------------------------------------------------------------------
template <bool HAS_CIN, bool HAS_BIAS, bool DO_TANH, bool OUT_BF16>
__global__ __launch_bounds__(256) void gemm_bf16_kernel(
    const unsigned short* __restrict__ A,
    const unsigned short* __restrict__ Bt,
    const unsigned short* __restrict__ Cin,   // bf16 [M,Nc] or null
    const float* __restrict__ bias,           // [Nc] or null
    void* __restrict__ Cout,                  // f32 or bf16 [M,Nc]
    int M, int Nc, int K) {
  const int tilesN = Nc >> 6;               // 64-wide N macro-tiles
  const int totalTiles = (M >> 5) * tilesN; // 32-tall M macro-tiles
  int tile = blockIdx.x * (blockDim.x >> 5) + (threadIdx.x >> 5);
  if (tile >= totalTiles) return;
  const int tm = tile / tilesN;
  const int tn = tile - tm * tilesN;
  const int lane = threadIdx.x & 31;
  const int l    = lane & 15;
  const int hi   = lane >> 4;      // half-wave select
  const int m0 = tm << 5, n0 = tn << 6;

  // ---- accumulator init (C layout: lanes 0-15 -> M 0..7, 16-31 -> M 8..15) --
  v8f acc[2][4];
#pragma unroll
  for (int mi = 0; mi < 2; ++mi) {
    const int crow = m0 + mi * 16 + (hi << 3);
#pragma unroll
    for (int ni = 0; ni < 4; ++ni) {
      const int ccol = n0 + ni * 16 + l;
      if (HAS_CIN) {
#pragma unroll
        for (int r = 0; r < 8; ++r)
          acc[mi][ni][r] = bf2f(Cin[(size_t)(crow + r) * Nc + ccol]);
      } else {
#pragma unroll
        for (int r = 0; r < 8; ++r) acc[mi][ni][r] = 0.0f;
      }
      if (HAS_BIAS) {
        float bv = bias[ccol];
#pragma unroll
        for (int r = 0; r < 8; ++r) acc[mi][ni][r] += bv;
      }
    }
  }

  // ---- K loop ---------------------------------------------------------------
  const unsigned short* arow[2];
  const unsigned short* brow[4];
#pragma unroll
  for (int mi = 0; mi < 2; ++mi) arow[mi] = A + (size_t)(m0 + mi * 16 + l) * K;
#pragma unroll
  for (int ni = 0; ni < 4; ++ni) brow[ni] = Bt + (size_t)(n0 + ni * 16 + l) * K;
  const int ak = hi ? 8 : 0;    // A 16x32 layout: halves K(0..7,16..23)/(8..15,24..31)
  const int bk = hi ? 8 : 0;    // B 32x16 layout: halves K(0..7,16..23)/(8..15,24..31)

  union V16U { v16bf v; v8bf h[2]; };
  for (int k0 = 0; k0 < K; k0 += 32) {
    if (k0 + 32 < K) {
      __builtin_prefetch(arow[0] + k0 + 32, 0, 1);   // global_prefetch_b8
      __builtin_prefetch(arow[1] + k0 + 32, 0, 1);
      __builtin_prefetch(brow[0] + k0 + 32, 0, 1);
      __builtin_prefetch(brow[2] + k0 + 32, 0, 1);
    }
    V16U av[2], bv[4];
#pragma unroll
    for (int mi = 0; mi < 2; ++mi) {
      av[mi].h[0] = *reinterpret_cast<const v8bf*>(arow[mi] + k0 + ak);
      av[mi].h[1] = *reinterpret_cast<const v8bf*>(arow[mi] + k0 + ak + 16);
    }
#pragma unroll
    for (int ni = 0; ni < 4; ++ni) {
      bv[ni].h[0] = *reinterpret_cast<const v8bf*>(brow[ni] + k0 + bk);
      bv[ni].h[1] = *reinterpret_cast<const v8bf*>(brow[ni] + k0 + bk + 16);
    }
#pragma unroll
    for (int mi = 0; mi < 2; ++mi)
#pragma unroll
      for (int ni = 0; ni < 4; ++ni)
        acc[mi][ni] = __builtin_amdgcn_wmma_f32_16x16x32_bf16(
            false, av[mi].v, false, bv[ni].v, (short)0, acc[mi][ni],
            false, false);
  }

  // ---- epilogue -------------------------------------------------------------
#pragma unroll
  for (int mi = 0; mi < 2; ++mi) {
    const int crow = m0 + mi * 16 + (hi << 3);
#pragma unroll
    for (int ni = 0; ni < 4; ++ni) {
      const int ccol = n0 + ni * 16 + l;
      if (OUT_BF16) {
        unsigned short* C = reinterpret_cast<unsigned short*>(Cout);
#pragma unroll
        for (int r = 0; r < 8; ++r) {
          float v = acc[mi][ni][r];
          if (DO_TANH) v = tanhf(v);
          C[(size_t)(crow + r) * Nc + ccol] = f2bf(v);
        }
      } else {
        float* C = reinterpret_cast<float*>(Cout);
#pragma unroll
        for (int r = 0; r < 8; ++r) {
          float v = acc[mi][ni][r];
          if (DO_TANH) v = tanhf(v);
          C[(size_t)(crow + r) * Nc + ccol] = v;
        }
      }
    }
  }
}

// ---------------------------------------------------------------------------
// LSTM gate nonlinearity: consumes G = xproj[t] + h@Whh^T (f32 [NS,4H]),
// updates c in place, emits next-h (bf16) and the timestep feature (f32+bf16).
// Gate order i,f,g,o (torch convention, matches reference split).
// ---------------------------------------------------------------------------
__global__ void lstm_gate_kernel(const float* __restrict__ G,
                                 float* __restrict__ c,
                                 unsigned short* __restrict__ h_bf16,
                                 float* __restrict__ feat,
                                 unsigned short* __restrict__ feat_bf16,
                                 int NS, int T, int t) {
  int idx = blockIdx.x * blockDim.x + threadIdx.x;   // NS*kH threads
  if (idx >= NS * kH) return;
  int n  = idx >> 9;          // /512
  int hh = idx & (kH - 1);
  const float* g = G + (size_t)n * kG4;
  float gi = sigmoidf_(g[hh]);
  float gf = sigmoidf_(g[kH + hh]);
  float gg = tanhf(g[2 * kH + hh]);
  float go = sigmoidf_(g[3 * kH + hh]);
  float cv = gf * c[idx] + gi * gg;
  c[idx] = cv;
  float hv = go * tanhf(cv);
  h_bf16[idx] = f2bf(hv);
  size_t fo = ((size_t)n * T + t) * kH + hh;
  feat[fo] = hv;
  feat_bf16[fo] = f2bf(hv);
}

// ---------------------------------------------------------------------------
// Attention scores: s[row] = dot(U[row,:], sw) + sb   (U already tanh'ed)
// one row per wave32, xor-shuffle reduce.
// ---------------------------------------------------------------------------
__global__ __launch_bounds__(256) void attn_score_kernel(
    const float* __restrict__ U, const float* __restrict__ sw,
    const float* __restrict__ sb, float* __restrict__ s, int rows) {
  int row = blockIdx.x * 8 + (threadIdx.x >> 5);
  if (row >= rows) return;
  int lane = threadIdx.x & 31;
  const float* u = U + (size_t)row * kH;
  float acc = 0.0f;
  for (int k = lane; k < kH; k += 32) acc += u[k] * sw[k];
#pragma unroll
  for (int off = 16; off > 0; off >>= 1) acc += __shfl_xor(acc, off, 32);
  if (lane == 0) s[row] = acc + sb[0];
}

// ---------------------------------------------------------------------------
// Masked softmax over time + weighted feature sum. One block per sequence.
// ---------------------------------------------------------------------------
__global__ __launch_bounds__(256) void attn_pool_kernel(
    const float* __restrict__ s, const float* __restrict__ feat,
    const int* __restrict__ lenA, const int* __restrict__ lenB, int NS1,
    int T, float* __restrict__ pooled) {
  int n = blockIdx.x;
  int len = (n < NS1) ? lenA[n] : lenB[n - NS1];
  __shared__ float red[256];
  __shared__ float w[256];          // T <= 256
  const float* sn = s + (size_t)n * T;
  int tid = threadIdx.x;

  float m = -1e30f;
  for (int t = tid; t < T; t += 256)
    if (t < len) m = fmaxf(m, sn[t]);
  red[tid] = m; __syncthreads();
  for (int st = 128; st > 0; st >>= 1) {
    if (tid < st) red[tid] = fmaxf(red[tid], red[tid + st]);
    __syncthreads();
  }
  m = red[0]; __syncthreads();

  float sum = 0.0f;
  for (int t = tid; t < T; t += 256) {
    float e = (t < len) ? __expf(sn[t] - m) : 0.0f;
    w[t] = e;
    sum += e;
  }
  red[tid] = sum; __syncthreads();
  for (int st = 128; st > 0; st >>= 1) {
    if (tid < st) red[tid] += red[tid + st];
    __syncthreads();
  }
  float inv = 1.0f / red[0];
  __syncthreads();

  for (int h = tid; h < kH; h += 256) {
    float acc = 0.0f;
    const float* fb = feat + (size_t)n * T * kH + h;
    for (int t = 0; t < len; ++t) acc += w[t] * fb[(size_t)t * kH];
    pooled[(size_t)n * kH + h] = acc * inv;
  }
}

// ---------------------------------------------------------------------------
// code_repr[b,h] = max_bl tanh(pooled[(b*8+bl), h])
// ---------------------------------------------------------------------------
__global__ void tanh_maxpool_kernel(const float* __restrict__ pooled,
                                    float* __restrict__ code) {
  int idx = blockIdx.x * blockDim.x + threadIdx.x;  // 32*512
  if (idx >= 32 * kH) return;
  int b = idx >> 9, h = idx & (kH - 1);
  float m = -1e30f;
#pragma unroll
  for (int bl = 0; bl < 8; ++bl)
    m = fmaxf(m, tanhf(pooled[(size_t)((b << 3) + bl) * kH + h]));
  code[idx] = m;
}

// ---------------------------------------------------------------------------
// Cosine sims + margin loss. Single wave32.
// docp holds 64 rows: 0..31 anchor pooled, 32..63 neg pooled.
// ---------------------------------------------------------------------------
__global__ void loss_kernel(const float* __restrict__ code,
                            const float* __restrict__ docp,
                            float* __restrict__ out) {
  int lane = threadIdx.x;   // blockDim = 32
  float loss = 0.0f;
  for (int b = 0; b < 32; ++b) {
    const float* c  = code + (size_t)b * kH;
    const float* a  = docp + (size_t)b * kH;
    const float* ng = docp + (size_t)(32 + b) * kH;
    float cc = 0, aa = 0, nn = 0, ca = 0, cn = 0;
    for (int k = lane; k < kH; k += 32) {
      float cv = c[k], av = a[k], nv = ng[k];
      cc += cv * cv; aa += av * av; nn += nv * nv;
      ca += cv * av; cn += cv * nv;
    }
#pragma unroll
    for (int off = 16; off > 0; off >>= 1) {
      cc += __shfl_xor(cc, off, 32);
      aa += __shfl_xor(aa, off, 32);
      nn += __shfl_xor(nn, off, 32);
      ca += __shfl_xor(ca, off, 32);
      cn += __shfl_xor(cn, off, 32);
    }
    const float eps = 1e-8f;
    float ncv = fmaxf(sqrtf(cc), eps);
    float simA = ca / (ncv * fmaxf(sqrtf(aa), eps));
    float simN = cn / (ncv * fmaxf(sqrtf(nn), eps));
    loss += fmaxf(0.05f - simA + simN, 1e-6f);
  }
  if (lane == 0) out[0] = loss / 32.0f;
}

// ---------------------------------------------------------------------------
// Host orchestration
// ---------------------------------------------------------------------------
extern "C" void kernel_launch(void* const* d_in, const int* in_sizes, int n_in,
                              void* d_out, int out_size, void* d_ws,
                              size_t ws_size, hipStream_t stream) {
  (void)in_sizes; (void)n_in; (void)out_size; (void)ws_size;

  // ---- inputs (setup_inputs dict order) ----
  const int*   tran           = (const int*)d_in[0];
  const int*   tran_block_len = (const int*)d_in[2];
  const int*   doc_anchor     = (const int*)d_in[3];
  const int*   doc_anchor_len = (const int*)d_in[4];
  const int*   doc_neg        = (const int*)d_in[5];
  const int*   doc_neg_len    = (const int*)d_in[6];
  const float* emb            = (const float*)d_in[7];
  const float* tran_wih       = (const float*)d_in[8];
  const float* tran_whh       = (const float*)d_in[9];
  const float* tran_b         = (const float*)d_in[10];
  const float* doc_wih        = (const float*)d_in[11];
  const float* doc_whh        = (const float*)d_in[12];
  const float* doc_b          = (const float*)d_in[13];
  const float* tran_attn_w    = (const float*)d_in[14];
  const float* tran_attn_b    = (const float*)d_in[15];
  const float* tran_s_w       = (const float*)d_in[16];
  const float* tran_s_b       = (const float*)d_in[17];
  const float* doc_attn_w     = (const float*)d_in[18];
  const float* doc_attn_b     = (const float*)d_in[19];
  const float* doc_s_w        = (const float*)d_in[20];
  const float* doc_s_b        = (const float*)d_in[21];

  // ---- workspace carve-out (256B aligned regions) ----
  char* wsp = (char*)d_ws;
  auto carve = [&](size_t bytes) -> void* {
    void* p = wsp;
    wsp += (bytes + 255) & ~(size_t)255;
    return p;
  };
  const int NS_T = 256, T_T = 128;   // tran: 32*8 sequences x 128
  const int NS_D = 64,  T_D = 256;   // docs: anchor+neg batched
  const size_t maxRows = (size_t)NS_T * T_T;   // 32768 >= 16384

  unsigned short* wih_t  = (unsigned short*)carve((size_t)kG4 * kE * 2);
  unsigned short* whh_t  = (unsigned short*)carve((size_t)kG4 * kH * 2);
  unsigned short* wih_d  = (unsigned short*)carve((size_t)kG4 * kE * 2);
  unsigned short* whh_d  = (unsigned short*)carve((size_t)kG4 * kH * 2);
  unsigned short* aw_t   = (unsigned short*)carve((size_t)kH * kH * 2);
  unsigned short* aw_d   = (unsigned short*)carve((size_t)kH * kH * 2);
  unsigned short* Xbf    = (unsigned short*)carve(maxRows * kE * 2);
  unsigned short* Xproj  = (unsigned short*)carve(maxRows * kG4 * 2);  // 128MB
  float*          Gbuf   = (float*)carve((size_t)NS_T * kG4 * 4);
  unsigned short* hbf    = (unsigned short*)carve((size_t)NS_T * kH * 2);
  float*          cbuf   = (float*)carve((size_t)NS_T * kH * 4);
  float*          feat   = (float*)carve(maxRows * kH * 4);            // 64MB
  unsigned short* featbf = (unsigned short*)carve(maxRows * kH * 2);   // 32MB
  float*          sbuf   = (float*)carve(maxRows * 4);
  float*          pool_t = (float*)carve((size_t)NS_T * kH * 4);
  float*          pool_d = (float*)carve((size_t)NS_D * kH * 4);
  float*          code   = (float*)carve((size_t)32 * kH * 4);
  float*          Ubuf   = (float*)Xproj;  // alias: Xproj dead once steps finish

  // ---- weight conversions ----
  auto cvt = [&](const float* src, unsigned short* dst, int n) {
    int n4 = n / 4;
    cvt_bf16_kernel<<<(n4 + 255) / 256, 256, 0, stream>>>(src, dst, n4);
  };
  cvt(tran_wih,    wih_t, kG4 * kE);
  cvt(tran_whh,    whh_t, kG4 * kH);
  cvt(doc_wih,     wih_d, kG4 * kE);
  cvt(doc_whh,     whh_d, kG4 * kH);
  cvt(tran_attn_w, aw_t,  kH * kH);
  cvt(doc_attn_w,  aw_d,  kH * kH);

  // ---- one LSTM + attention-pool pipeline ----
  auto run_phase = [&](int NS, int T,
                       const int* tokA, const int* tokB, int NS1tok,
                       const int* lenA, const int* lenB, int NS1len,
                       const unsigned short* wih, const unsigned short* whh,
                       const float* bias, const unsigned short* aw,
                       const float* ab, const float* swv, const float* sbv,
                       float* pooled) {
    const int rows = NS * T;
    // 1) embedding gather -> time-major bf16 X
    gather_embed_kernel<<<rows, 128, 0, stream>>>(tokA, tokB, NS1tok, NS, T,
                                                  emb, Xbf);
    // 2) hoisted input projection: Xproj = X @ wih^T + b   (bf16 out)
    {
      int tiles = (rows / 32) * (kG4 / 64);
      gemm_bf16_kernel<false, true, false, true>
          <<<tiles / 8, 256, 0, stream>>>(Xbf, wih, nullptr, bias,
                                          (void*)Xproj, rows, kG4, kE);
    }
    // 3) recurrence
    hipMemsetAsync(cbuf, 0, (size_t)NS * kH * sizeof(float), stream);
    hipMemsetAsync(hbf,  0, (size_t)NS * kH * sizeof(unsigned short), stream);
    const int stepTiles = (NS / 32) * (kG4 / 64);
    for (int t = 0; t < T; ++t) {
      gemm_bf16_kernel<true, false, false, false>
          <<<stepTiles / 8, 256, 0, stream>>>(
              hbf, whh, Xproj + (size_t)t * NS * kG4, nullptr, (void*)Gbuf,
              NS, kG4, kH);
      lstm_gate_kernel<<<(NS * kH) / 256, 256, 0, stream>>>(
          Gbuf, cbuf, hbf, feat, featbf, NS, T, t);
    }
    // 4) U = tanh(feat @ aw^T + ab)   (f32 out, aliases Xproj region)
    {
      int tiles = (rows / 32) * (kH / 64);
      gemm_bf16_kernel<false, true, true, false>
          <<<tiles / 8, 256, 0, stream>>>(featbf, aw, nullptr, ab,
                                          (void*)Ubuf, rows, kH, kH);
    }
    // 5) scores, masked softmax, weighted sum
    attn_score_kernel<<<(rows + 7) / 8, 256, 0, stream>>>(Ubuf, swv, sbv,
                                                          sbuf, rows);
    attn_pool_kernel<<<NS, 256, 0, stream>>>(sbuf, feat, lenA, lenB, NS1len,
                                             T, pooled);
  };

  // tran path: 256 sub-sequences of length 128
  run_phase(NS_T, T_T, tran, tran, NS_T,
            tran_block_len, tran_block_len, NS_T,
            wih_t, whh_t, tran_b, aw_t, tran_attn_b, tran_s_w, tran_s_b,
            pool_t);
  // doc path: anchor (rows 0..31) + neg (rows 32..63) batched
  run_phase(NS_D, T_D, doc_anchor, doc_neg, 32,
            doc_anchor_len, doc_neg_len, 32,
            wih_d, whh_d, doc_b, aw_d, doc_attn_b, doc_s_w, doc_s_b,
            pool_d);

  // code_repr = max over block of tanh(pooled)
  tanh_maxpool_kernel<<<(32 * kH) / 256, 256, 0, stream>>>(pool_t, code);
  // cosine sims + clipped margin loss
  loss_kernel<<<1, 32, 0, stream>>>(code, pool_d, (float*)d_out);
}
